// Convolution_22917945491536
// MI455X (gfx1250) — matrix operand order repaired and checked by reading, not compiled, not run on hardware
//
#include <hip/hip_runtime.h>

// ---------------------------------------------------------------------------
// GNN message-passing layer on MI455X (gfx1250, wave32, WMMA).
//
// Roofline: edge GEMM = 52.4 GFLOP dominates; x (25.6MB) + W (128KB) fit in
// the 192MB L2, so gathers/scatters are L2-bound, HBM traffic ~100MB
// (~4us @ 23.3 TB/s).  => both GEMMs run on v_wmma_f32_16x16x32_bf16
// (f32 accumulate).  v2: persistent blocks stage the packed weights in LDS
// (64KB / 32KB of the 320KB per WGP) so B fragments come from ds_load_b128
// instead of re-streaming through the vmem path every tile; the vmem path is
// left for the random x[src]/x[dst] gathers and the f32 atomic scatter.
// ---------------------------------------------------------------------------

#define H 128
#define EDGE_WAVES 8   // 256-thread blocks

typedef __attribute__((ext_vector_type(16))) __bf16 v16bf;
typedef __attribute__((ext_vector_type(8)))  float  v8f;

union Frag16 { v16bf v; uint4 q[2]; };

__device__ __forceinline__ unsigned short f2bf_rne(float f) {
  unsigned int u = __float_as_uint(f);
  unsigned int r = u + 0x7FFFu + ((u >> 16) & 1u);   // round-to-nearest-even
  return (unsigned short)(r >> 16);
}

// ---------------------------------------------------------------------------
// Pack W[k][n] (row-major, n-dim = 128) into per-lane WMMA B fragments.
// Fragment (nt, kc): lane L holds column nt*16 + (L&15), K = kc*32 + (L>>4)*8 + {0..7, 16..23}.
// Dst index:  frag*512 + lane*16  ==  t*16  for t = frag*32 + lane.
// ---------------------------------------------------------------------------
__global__ void pack_w_kernel(const float* __restrict__ W,
                              unsigned short* __restrict__ Wp,
                              int kch, int total_threads) {
  int t = blockIdx.x * blockDim.x + threadIdx.x;
  if (t >= total_threads) return;
  int lane = t & 31;
  int frag = t >> 5;
  int kc   = frag % kch;
  int nt   = frag / kch;
  int n  = nt * 16 + (lane & 15);
  int k0 = kc * 32 + (lane >> 4) * 8;
  unsigned short* dst = Wp + (size_t)t * 16;
#pragma unroll
  for (int j = 0; j < 16; ++j)
    dst[j] = f2bf_rne(W[(size_t)(k0 + j + (j >= 8 ? 8 : 0)) * H + n]);
}

__global__ void f32_to_bf16_kernel(const float* __restrict__ x,
                                   unsigned short* __restrict__ xb, int n4) {
  int i = blockIdx.x * blockDim.x + threadIdx.x;
  if (i >= n4) return;
  float4 f = ((const float4*)x)[i];
  ushort4 o;
  o.x = f2bf_rne(f.x); o.y = f2bf_rne(f.y);
  o.z = f2bf_rne(f.z); o.w = f2bf_rne(f.w);
  ((ushort4*)xb)[i] = o;
}

__global__ void zero_kernel(float* __restrict__ p, int n4) {
  int i = blockIdx.x * blockDim.x + threadIdx.x;
  if (i >= n4) return;
  ((float4*)p)[i] = make_float4(0.f, 0.f, 0.f, 0.f);
}

// t = bf16(x + scale * m_acc)   (scale = 1/sqrt(avg_degree) = 0.25)
__global__ void residual_bf16_kernel(const float* __restrict__ x,
                                     const float* __restrict__ macc,
                                     unsigned short* __restrict__ tb,
                                     int n4, float scale) {
  int i = blockIdx.x * blockDim.x + threadIdx.x;
  if (i >= n4) return;
  float4 a = ((const float4*)x)[i];
  float4 m = ((const float4*)macc)[i];
  ushort4 o;
  o.x = f2bf_rne(fmaf(scale, m.x, a.x));
  o.y = f2bf_rne(fmaf(scale, m.y, a.y));
  o.z = f2bf_rne(fmaf(scale, m.z, a.z));
  o.w = f2bf_rne(fmaf(scale, m.w, a.w));
  ((ushort4*)tb)[i] = o;
}

// ---------------------------------------------------------------------------
// Edge kernel: persistent 8-wave blocks; packed W_interact (64KB) staged in
// LDS once per block; each wave grid-strides over 16-edge tiles.
//   A = bf16(concat(x[src], x[dst]))  : 16 x 256   (gathers via vmem)
//   m = relu(A @ W_interact + b)      : 16 x 128   (64 WMMAs, B via ds_load)
//   scatter: m_acc[src] += m          (hardware f32 atomic adds, land in L2)
// ---------------------------------------------------------------------------
__global__ __launch_bounds__(256)
void edge_msg_kernel(const unsigned short* __restrict__ xb,
                     const unsigned short* __restrict__ Wbi,
                     const long long* __restrict__ esrc,
                     const long long* __restrict__ edst,
                     const float* __restrict__ b_interact,
                     float* __restrict__ macc, int ntiles) {
  __shared__ unsigned short sW[2 * H * H];          // 64KB of 320KB/WGP
  {
    const uint4* g = (const uint4*)Wbi;
    uint4*       s = (uint4*)sW;
    int tid = threadIdx.x;
#pragma unroll
    for (int i = 0; i < (2 * H * H) / (8 * 256); ++i)   // 4096 uint4 / 256 thr
      s[tid + i * 256] = g[tid + i * 256];
  }
  __syncthreads();

  int lane   = threadIdx.x & 31;
  int waveid = threadIdx.x >> 5;
  int half   = lane >> 4;
  int m      = lane & 15;

  // bias depends only on (nt, lane): hoist out of the tile loop
  float bias_r[8];
#pragma unroll
  for (int nt = 0; nt < 8; ++nt) bias_r[nt] = b_interact[nt * 16 + m];

  for (int tile = blockIdx.x * EDGE_WAVES + waveid; tile < ntiles;
       tile += gridDim.x * EDGE_WAVES) {             // wave-uniform bound
    int eb = tile * 16;
    const unsigned short* sp = xb + (size_t)((int)esrc[eb + m]) * H;
    const unsigned short* dp = xb + (size_t)((int)edst[eb + m]) * H;

    Frag16 a[8];
#pragma unroll
    for (int kc = 0; kc < 8; ++kc) {
      const unsigned short* base = (kc < 4) ? (sp + kc * 32) : (dp + (kc - 4) * 32);
      a[kc].q[0] = *(const uint4*)(base + half * 8);
      a[kc].q[1] = *(const uint4*)(base + 16 + half * 8);
    }

    // D layout: VGPR r of this lane holds row (r + 8*half) of the tile.
    int drow[8];
#pragma unroll
    for (int r = 0; r < 8; ++r) drow[r] = (int)esrc[eb + r + 8 * half];

#pragma unroll
    for (int nt = 0; nt < 8; ++nt) {
      v8f acc = {};
#pragma unroll
      for (int kc = 0; kc < 8; ++kc) {
        Frag16 b;
        const unsigned short* bp = sW + (nt * 8 + kc) * 512 + lane * 16;
        b.q[0] = ((const uint4*)bp)[0];
        b.q[1] = ((const uint4*)bp)[1];
        acc = __builtin_amdgcn_wmma_f32_16x16x32_bf16(
            false, a[kc].v, false, b.v, (short)0, acc, false, false);
      }
      int ncol = nt * 16 + m;
      float bias = bias_r[nt];
#pragma unroll
      for (int r = 0; r < 8; ++r) {
        float v = acc[r] + bias;
        v = v > 0.f ? v : 0.f;
        (void)__hip_atomic_fetch_add(macc + (size_t)drow[r] * H + ncol, v,
                                     __ATOMIC_RELAXED, __HIP_MEMORY_SCOPE_AGENT);
      }
    }
  }
}

// ---------------------------------------------------------------------------
// Update kernel: persistent 8-wave blocks; packed W_update (32KB) staged in
// LDS; out = x + relu(tb @ W_update + b_update), tb = bf16(x + 0.25*m_acc).
// ---------------------------------------------------------------------------
__global__ __launch_bounds__(256)
void node_update_kernel(const unsigned short* __restrict__ tb,
                        const unsigned short* __restrict__ Wbu,
                        const float* __restrict__ x,
                        const float* __restrict__ b_update,
                        float* __restrict__ out, int ntiles) {
  __shared__ unsigned short sW[H * H];              // 32KB
  {
    const uint4* g = (const uint4*)Wbu;
    uint4*       s = (uint4*)sW;
    int tid = threadIdx.x;
#pragma unroll
    for (int i = 0; i < (H * H) / (8 * 256); ++i)   // 2048 uint4 / 256 thr
      s[tid + i * 256] = g[tid + i * 256];
  }
  __syncthreads();

  int lane   = threadIdx.x & 31;
  int waveid = threadIdx.x >> 5;
  int half   = lane >> 4;
  int m      = lane & 15;

  float bias_r[8];
#pragma unroll
  for (int nt = 0; nt < 8; ++nt) bias_r[nt] = b_update[nt * 16 + m];

  for (int tile = blockIdx.x * EDGE_WAVES + waveid; tile < ntiles;
       tile += gridDim.x * EDGE_WAVES) {
    int nb = tile * 16;
    const unsigned short* tp = tb + (size_t)(nb + m) * H;

    Frag16 a[4];
#pragma unroll
    for (int kc = 0; kc < 4; ++kc) {
      const unsigned short* base = tp + kc * 32;
      a[kc].q[0] = *(const uint4*)(base + half * 8);
      a[kc].q[1] = *(const uint4*)(base + 16 + half * 8);
    }

    int row0 = nb + 8 * half;
#pragma unroll
    for (int nt = 0; nt < 8; ++nt) {
      v8f acc = {};
#pragma unroll
      for (int kc = 0; kc < 4; ++kc) {
        Frag16 b;
        const unsigned short* bp = sW + (nt * 4 + kc) * 512 + lane * 16;
        b.q[0] = ((const uint4*)bp)[0];
        b.q[1] = ((const uint4*)bp)[1];
        acc = __builtin_amdgcn_wmma_f32_16x16x32_bf16(
            false, a[kc].v, false, b.v, (short)0, acc, false, false);
      }
      int ncol = nt * 16 + m;
      float bias = bias_r[nt];
#pragma unroll
      for (int r = 0; r < 8; ++r) {
        float v = acc[r] + bias;
        v = v > 0.f ? v : 0.f;
        size_t o = (size_t)(row0 + r) * H + ncol;
        out[o] = x[o] + v;
      }
    }
  }
}

// ---------------------------------------------------------------------------
extern "C" void kernel_launch(void* const* d_in, const int* in_sizes, int n_in,
                              void* d_out, int out_size, void* d_ws, size_t ws_size,
                              hipStream_t stream) {
  const float*     x    = (const float*)d_in[0];
  const long long* eidx = (const long long*)d_in[1];   // int64 [2, E]
  const float*     Wi   = (const float*)d_in[2];       // [2H, H]
  const float*     bi   = (const float*)d_in[3];
  const float*     Wu   = (const float*)d_in[4];       // [H, H]
  const float*     bu   = (const float*)d_in[5];
  float*           out  = (float*)d_out;

  const int NH = in_sizes[0];          // N*H
  const int N  = NH / H;               // 50000
  const int E  = in_sizes[1] / 2;      // 800000
  const long long* esrc = eidx;
  const long long* edst = eidx + E;

  // d_ws layout (all 256B-aligned offsets)
  char* ws = (char*)d_ws;
  float*          macc = (float*)ws;                              // N*H f32
  unsigned short* xb   = (unsigned short*)(ws + (size_t)NH * 4);  // N*H bf16
  unsigned short* Wbi  = (unsigned short*)(ws + (size_t)NH * 6);  // 2H*H bf16 packed
  unsigned short* Wbu  = Wbi + 2 * H * H;                         // H*H bf16 packed

  // 1) pack weights into WMMA B-fragment layout
  pack_w_kernel<<<(8 * 8 * 32 + 255) / 256, 256, 0, stream>>>(Wi, Wbi, 8, 8 * 8 * 32);
  pack_w_kernel<<<(8 * 4 * 32 + 255) / 256, 256, 0, stream>>>(Wu, Wbu, 4, 8 * 4 * 32);

  // 2) x -> bf16, zero accumulator
  int n4 = NH / 4;
  f32_to_bf16_kernel<<<(n4 + 255) / 256, 256, 0, stream>>>(x, xb, n4);
  zero_kernel<<<(n4 + 255) / 256, 256, 0, stream>>>(macc, n4);

  // 3) edge GEMM + ReLU + atomic scatter (persistent blocks, grid-stride)
  int etiles = E / 16;                                  // 50000
  int eblocks = (etiles + EDGE_WAVES - 1) / EDGE_WAVES;
  if (eblocks > 2048) eblocks = 2048;
  edge_msg_kernel<<<eblocks, 256, 0, stream>>>(xb, Wbi, esrc, edst, bi,
                                               macc, etiles);

  // 4) t = bf16(x + 0.25 * m_acc)  (reuses xb buffer)
  residual_bf16_kernel<<<(n4 + 255) / 256, 256, 0, stream>>>(x, macc, xb, n4, 0.25f);

  // 5) update GEMM + ReLU + residual (persistent blocks, grid-stride)
  int ntilesN = N / 16;                                 // 3125
  int ublocks = (ntilesN + EDGE_WAVES - 1) / EDGE_WAVES;
  if (ublocks > 1024) ublocks = 1024;
  node_update_kernel<<<ublocks, 256, 0, stream>>>(xb, Wbu, x, bu, out, ntilesN);
}
